// SensorEmbedder_49435073577257
// MI455X (gfx1250) — compile-verified, matrix operand order checked by reading, not run on hardware
//
#include <hip/hip_runtime.h>

typedef __attribute__((ext_vector_type(16))) __bf16 v16bf;
typedef __attribute__((ext_vector_type(8)))  __bf16 v8bf;
typedef __attribute__((ext_vector_type(8)))  float  v8f;

namespace {
constexpr int kNObs   = 2000000;
constexpr int kMeta   = 28;
constexpr int kEmb    = 4;
constexpr int kHid    = 64;
constexpr int kMlpIn  = 33;
constexpr int kMlpOut = 31;
constexpr int kSed    = 32;
constexpr int kOut    = 512;
constexpr int kNPix   = 16384;
constexpr int kNBatch = 8;
constexpr int kNBuck  = 128;
constexpr int kNSeg   = kNBatch * kNPix;   // 131072
constexpr float kLnEps = 1e-5f;
constexpr int kWaves  = 8;                 // waves per block (wave32)
constexpr int kBlk    = kWaves * 32;       // 256 threads
// bf16 row strides: must be multiple of 8 (16B b128 alignment) and give
// distinct banks across 16 lanes: stride*2B/4 mod 64 coprime-ish with 64.
constexpr int kStrA   = 88;                // 176B rows: 44n mod 64 distinct
constexpr int kStrP   = 40;                // 80B rows: 20n mod 64 distinct
}

union FragU { v16bf v; v8bf h[2]; };

// A lane's 16 fragment elements are two contiguous K-runs of 8 bf16:
// e=0..7 -> k = g*8 + e ; e=8..15 -> k = 16 + g*8 + (e-8)   (+32 per chunk)
__device__ __forceinline__ v16bf load_frag(const __bf16* rowp, int off0) {
  FragU u;
  u.h[0] = *(const v8bf*)(rowp + off0);
  u.h[1] = *(const v8bf*)(rowp + off0 + 16);
  return u.v;
}

__device__ __forceinline__ v8f wmma_bf16(v16bf a, v16bf b, v8f c) {
  return __builtin_amdgcn_wmma_f32_16x16x32_bf16(
      false, a, false, b, (short)0, c, false, false);
}

__device__ __forceinline__ void atomic_add_f32(float* p, float v) {
  (void)__hip_atomic_fetch_add(p, v, __ATOMIC_RELAXED, __HIP_MEMORY_SCOPE_AGENT);
}

// ---------------------------------------------------------------------------
// Kernel A: tokenizer MLP (WMMA bf16) + bucket add + segment-sum scatter.
// One wave handles 16 consecutive observations.
// ---------------------------------------------------------------------------
__global__ __launch_bounds__(kBlk) void tokenize_scatter(
    const float* __restrict__ obs, const float* __restrict__ md,
    const int* __restrict__ pix, const int* __restrict__ lch,
    const int* __restrict__ lpl, const int* __restrict__ otype,
    const int* __restrict__ offs, const float* __restrict__ embed,
    const float* __restrict__ w1, const float* __restrict__ b1,
    const float* __restrict__ lng, const float* __restrict__ lnb,
    const float* __restrict__ w2, const float* __restrict__ b2,
    const float* __restrict__ bkt,
    float* __restrict__ sums, float* __restrict__ cnts) {
  __shared__ alignas(16) __bf16 sW1T[64 * kStrA];  // [n][k] transposed, 11 KB
  __shared__ alignas(16) __bf16 sW2T[32 * kStrA];  // [j][k] transposed, 5.5 KB
  __shared__ alignas(16) __bf16 sXb[kWaves][16 * kStrA];  // x rows / h rows
  __shared__ float sB1[64], sLG[64], sLB[64], sB2[32];
  __shared__ float sBkt[kNBuck * kSed];            // 16 KB
  __shared__ int   sSeg[kWaves][16];
  __shared__ int   sBid[kWaves][16];
  __shared__ float sObs[kWaves][16];

  const int tid = threadIdx.x;
  // Stage transposed bf16 weights, K zero-padded to 64.
  for (int idx = tid; idx < 64 * 64; idx += kBlk) {
    int n = idx >> 6, k = idx & 63;
    sW1T[n * kStrA + k] = (__bf16)((k < kMlpIn) ? w1[k * kHid + n] : 0.f);
  }
  for (int idx = tid; idx < 32 * 64; idx += kBlk) {
    int j = idx >> 6, k = idx & 63;
    sW2T[j * kStrA + k] = (__bf16)((j < kMlpOut) ? w2[k * kMlpOut + j] : 0.f);
  }
  if (tid < 64) { sB1[tid] = b1[tid]; sLG[tid] = lng[tid]; sLB[tid] = lnb[tid]; }
  if (tid < 32) sB2[tid] = (tid < kMlpOut) ? b2[tid] : 0.f;
  for (int idx = tid; idx < kNBuck * kSed; idx += kBlk) sBkt[idx] = bkt[idx];

  const int wave = tid >> 5, lane = tid & 31;
  const int g = lane >> 4, ln = lane & 15;
  const int base = (blockIdx.x * kWaves + wave) * 16;
  const int row = base + ln;
  __bf16* xw = sXb[wave];

  if (g == 0) {
    int bi = 0;
    #pragma unroll
    for (int j = 0; j < kNBatch; j++) bi += (offs[j] <= row) ? 1 : 0;
    if (bi > kNBatch - 1) bi = kNBatch - 1;
    sSeg[wave][ln] = bi * kNPix + pix[row];
    sBid[wave][ln] = lpl[row] * 16 + lch[row];
    float o = obs[row];
    sObs[wave][ln] = o;
    xw[ln * kStrA + 0] = (__bf16)o;                 // x_in col 0
  }
  // metadata for the wave's 16 rows is one contiguous 448-f32 block: load
  // coalesced and scatter to bf16 LDS rows (cols 1..28).
  {
    const size_t mb = (size_t)base * kMeta;
    #pragma unroll
    for (int it = 0; it < 14; it++) {
      int idx = it * 32 + lane;                     // 448 = 14*32
      int r = idx / kMeta, k = idx - r * kMeta;
      xw[r * kStrA + 1 + k] = (__bf16)md[mb + idx];
    }
  }
  // obs-type embedding gather (4 cols) by high half-wave; zero-pad k=33..63.
  if (g == 1) {
    const int ot = otype[row];
    #pragma unroll
    for (int e = 0; e < kEmb; e++)
      xw[ln * kStrA + 29 + e] = (__bf16)embed[ot * kEmb + e];
  }
  {
    int k0 = 33 + g * 16;                           // g0: 33..48, g1: 49..63
    int k1 = (g == 0) ? 49 : 64;
    for (int k = k0; k < k1; k++) xw[ln * kStrA + k] = (__bf16)0.f;
  }
  __syncthreads();

  // ---- GEMM1: h(16x64) = x_in @ w1 + b1 ; 2 K-chunks x 4 N-tiles ----
  const int offA = g * 8;
  const v16bf a0 = load_frag(xw + ln * kStrA, offA);
  const v16bf a1 = load_frag(xw + ln * kStrA, offA + 32);
  v8f acc[4];
  #pragma unroll
  for (int t = 0; t < 4; t++) {
    const int n = t * 16 + ln;
    const v16bf b0 = load_frag(sW1T + n * kStrA, offA);
    const v16bf b1v = load_frag(sW1T + n * kStrA, offA + 32);
    v8f c;
    #pragma unroll
    for (int v = 0; v < 8; v++) c[v] = sB1[n];
    c = wmma_bf16(a0, b0, c);
    c = wmma_bf16(a1, b1v, c);
    acc[t] = c;
  }

  // ---- LayerNorm(64) + SiLU on C layout (row M = one VGPR x 16 lanes) ----
  #pragma unroll
  for (int v = 0; v < 8; v++) {
    float s = 0.f, q = 0.f;
    #pragma unroll
    for (int t = 0; t < 4; t++) { float x = acc[t][v]; s += x; q += x * x; }
    #pragma unroll
    for (int m = 1; m <= 8; m <<= 1) {
      s += __shfl_xor(s, m, 32);
      q += __shfl_xor(q, m, 32);
    }
    const float mean = s * (1.f / 64.f);
    const float var  = q * (1.f / 64.f) - mean * mean;
    const float inv  = rsqrtf(var + kLnEps);
    #pragma unroll
    for (int t = 0; t < 4; t++) {
      const int n = t * 16 + ln;
      float x = (acc[t][v] - mean) * inv * sLG[n] + sLB[n];
      x = x * (1.f / (1.f + __expf(-x)));           // SiLU
      acc[t][v] = x;
    }
  }

  // ---- transpose h back into bf16 rows (reuse sXb) ----
  __syncthreads();
  #pragma unroll
  for (int t = 0; t < 4; t++)
    #pragma unroll
    for (int v = 0; v < 8; v++)
      xw[(v + 8 * g) * kStrA + t * 16 + ln] = (__bf16)acc[t][v];
  __syncthreads();

  // ---- GEMM2: mlp_out(16x31) = h(16x64) @ w2 + b2 ----
  const v16bf ha0 = load_frag(xw + ln * kStrA, offA);
  const v16bf ha1 = load_frag(xw + ln * kStrA, offA + 32);
  v8f acc2[2];
  #pragma unroll
  for (int t = 0; t < 2; t++) {
    const int j = t * 16 + ln;
    const v16bf b0 = load_frag(sW2T + j * kStrA, offA);
    const v16bf b1v = load_frag(sW2T + j * kStrA, offA + 32);
    v8f c;
    #pragma unroll
    for (int v = 0; v < 8; v++) c[v] = sB2[j];
    c = wmma_bf16(ha0, b0, c);
    c = wmma_bf16(ha1, b1v, c);
    acc2[t] = c;
  }

  // ---- scatter: feat = [obs, mlp_out] + bucket_embed[bid] ; L2 atomics ----
  #pragma unroll
  for (int t = 0; t < 2; t++)
    #pragma unroll
    for (int v = 0; v < 8; v++) {
      const int c = t * 16 + ln + 1;     // encoded column (col 0 = raw obs)
      if (c < kSed) {
        const int M = v + 8 * g;
        const int sg = sSeg[wave][M];
        const float val = acc2[t][v] + sBkt[sBid[wave][M] * kSed + c];
        atomic_add_f32(&sums[(size_t)sg * kSed + c], val);
      }
    }
  if (ln == 0) {
    #pragma unroll
    for (int v = 0; v < 8; v++) {
      const int M = v + 8 * g;
      const float val = sObs[wave][M] + sBkt[sBid[wave][M] * kSed];
      atomic_add_f32(&sums[(size_t)sSeg[wave][M] * kSed], val);
    }
  }
  if (g == 0) atomic_add_f32(&cnts[sSeg[wave][ln]], 1.0f);
}

// ---------------------------------------------------------------------------
// Kernel B: segment mean / infill, then (nseg x 32) @ wp(32x512) + bp.
// One wave = one 16-segment M-tile; 32 N-tiles of 16 via WMMA.
// ---------------------------------------------------------------------------
__global__ __launch_bounds__(kBlk) void finalize_project(
    const float* __restrict__ sums, const float* __restrict__ cnts,
    const float* __restrict__ infill, const float* __restrict__ wp,
    const float* __restrict__ bp, float* __restrict__ out) {
  __shared__ alignas(16) __bf16 sWpT[kOut * kStrP];       // [n][k], 40 KB
  __shared__ alignas(16) __bf16 sCell[kWaves][16 * kStrP];
  __shared__ float sBp[kOut];
  __shared__ float sInf[kSed];

  const int tid = threadIdx.x;
  for (int idx = tid; idx < kSed * kOut; idx += kBlk) {
    int k = idx >> 9, n = idx & 511;                // coalesced wp read
    sWpT[n * kStrP + k] = (__bf16)wp[idx];
  }
  for (int idx = tid; idx < kOut; idx += kBlk) sBp[idx] = bp[idx];
  if (tid < kSed) sInf[tid] = infill[tid];

  const int wave = tid >> 5, lane = tid & 31;
  const int g = lane >> 4, ln = lane & 15;
  const int m0 = (blockIdx.x * kWaves + wave) * 16;
  __bf16* cw = sCell[wave];
  __syncthreads();

  {  // cell = cnts>0 ? sums/cnts : infill  (half-wave per 16 columns)
    const size_t s = (size_t)(m0 + ln);
    const float cnt = cnts[s];
    const float rc = (cnt > 0.f) ? (1.f / cnt) : 0.f;
    const float* srow = sums + s * kSed + g * 16;
    #pragma unroll
    for (int k = 0; k < 16; k++) {
      const int c = g * 16 + k;
      cw[ln * kStrP + c] = (__bf16)((cnt > 0.f) ? (srow[k] * rc) : sInf[c]);
    }
  }
  __syncthreads();

  const v16bf a = load_frag(cw + ln * kStrP, g * 8);
  #pragma unroll 4
  for (int t = 0; t < 32; t++) {
    const int N = t * 16 + ln;
    const v16bf b = load_frag(sWpT + N * kStrP, g * 8);
    v8f c;
    #pragma unroll
    for (int v = 0; v < 8; v++) c[v] = sBp[N];
    c = wmma_bf16(a, b, c);
    #pragma unroll
    for (int v = 0; v < 8; v++)
      out[(size_t)(m0 + v + 8 * g) * kOut + N] = c[v];
  }
}

extern "C" void kernel_launch(void* const* d_in, const int* in_sizes, int n_in,
                              void* d_out, int out_size, void* d_ws,
                              size_t ws_size, hipStream_t stream) {
  const float* obs    = (const float*)d_in[0];
  const float* md     = (const float*)d_in[1];
  const int*   pix    = (const int*)d_in[2];
  const int*   lch    = (const int*)d_in[3];
  const int*   lpl    = (const int*)d_in[4];
  const int*   otype  = (const int*)d_in[5];
  const int*   offs   = (const int*)d_in[6];
  // d_in[7] = npix (compile-time constant kNPix)
  const float* embed  = (const float*)d_in[8];
  const float* w1     = (const float*)d_in[9];
  const float* b1     = (const float*)d_in[10];
  const float* lng    = (const float*)d_in[11];
  const float* lnb    = (const float*)d_in[12];
  const float* w2     = (const float*)d_in[13];
  const float* b2     = (const float*)d_in[14];
  const float* bkt    = (const float*)d_in[15];
  const float* infill = (const float*)d_in[16];
  const float* wp     = (const float*)d_in[17];
  const float* bp     = (const float*)d_in[18];

  float* sums = (float*)d_ws;
  float* cnts = sums + (size_t)kNSeg * kSed;

  hipMemsetAsync(d_ws, 0, (size_t)kNSeg * (kSed + 1) * sizeof(float), stream);

  dim3 blk(kBlk);
  dim3 grdA(kNObs / (16 * kWaves));  // 15625 blocks, 1 wave = 16 obs
  tokenize_scatter<<<grdA, blk, 0, stream>>>(obs, md, pix, lch, lpl, otype,
                                             offs, embed, w1, b1, lng, lnb, w2,
                                             b2, bkt, sums, cnts);

  dim3 grdB(kNSeg / (16 * kWaves));  // 1024 blocks, 1 wave = 16 segments
  finalize_project<<<grdB, blk, 0, stream>>>(sums, cnts, infill, wp, bp,
                                             (float*)d_out);
}
